// Transformer_26182120636782
// MI455X (gfx1250) — matrix operand order repaired and checked by reading, not compiled, not run on hardware
//
#include <hip/hip_runtime.h>
#include <cstdint>
#include <cstddef>

#define DEV __device__ __forceinline__

typedef __bf16 bf16x16 __attribute__((ext_vector_type(16)));
typedef float  f32x8   __attribute__((ext_vector_type(8)));
typedef unsigned int u32x4 __attribute__((ext_vector_type(4)));
typedef unsigned short u16;

// Problem dims (fixed by the reference)
constexpr int Bc   = 8;
constexpr int Lc   = 1024;
constexpr int Dc   = 512;
constexpr int Hc   = 8;
constexpr int DKc  = 64;
constexpr int DFFc = 1024;
constexpr int NLAY = 4;
constexpr int MR   = Bc * Lc;      // 8192 rows

DEV u16 f2b(float f) {
  unsigned int u = __float_as_uint(f);
  u += 0x7FFFu + ((u >> 16) & 1u);      // round-to-nearest-even
  return (u16)(u >> 16);
}

union FragU { bf16x16 v; u32x4 q[2]; };

// A-fragment (16x32 bf16, MxK): lane m holds row m, K in {g*8..g*8+7, g*8+16..g*8+23}
DEV bf16x16 ldsA(const u16* rowp, int lane) {
  int g = (lane >> 4) & 1;
  FragU u;
  u.q[0] = *(const u32x4*)(rowp + g * 8);
  u.q[1] = *(const u32x4*)(rowp + g * 8 + 16);
  return u.v;
}
// B-fragment (32x16 bf16, KxN), B stored transposed (N rows of K): lane n holds
// column n, K in {g*8..g*8+7, g*8+16..g*8+23}
DEV bf16x16 ldsB(const u16* rowp, int lane) {
  int g = (lane >> 4) & 1;
  FragU u;
  u.q[0] = *(const u32x4*)(rowp + g * 8);
  u.q[1] = *(const u32x4*)(rowp + g * 8 + 16);
  return u.v;
}

DEV f32x8 wmma_bf16(bf16x16 a, bf16x16 b, f32x8 c) {
  return __builtin_amdgcn_wmma_f32_16x16x32_bf16(false, a, false, b, (short)0, c,
                                                 false, false);
}

// ---------------------------------------------------------------- converters
__global__ void cvt_bf16(const float* __restrict__ s, u16* __restrict__ d, int n) {
  int i = blockIdx.x * blockDim.x + threadIdx.x;
  if (i < n) d[i] = f2b(s[i]);
}

// Wq/Wk/Wv: (NLAY,H,D,DK) -> combined per-layer (1536 rows, K=D) transposed bf16.
// rowOff selects the q (0), k (512), v (1024) row band.
__global__ void cvt_wqkvT(const float* __restrict__ W, u16* __restrict__ Wt,
                          int rowOff) {
  int i = blockIdx.x * blockDim.x + threadIdx.x;
  if (i >= NLAY * Hc * Dc * DKc) return;
  int e = i & 63;  int t = i >> 6;
  int d = t & 511; t >>= 9;
  int h = t & 7;   int lay = t >> 3;
  int n = rowOff + h * 64 + e;
  Wt[((size_t)lay * 1536 + n) * 512 + d] = f2b(W[i]);
}

// generic (NLAY,K,N) -> per-layer (N rows, K) transposed bf16
__global__ void cvt_wT(const float* __restrict__ W, u16* __restrict__ Wt, int K, int N) {
  int i = blockIdx.x * blockDim.x + threadIdx.x;
  if (i >= NLAY * K * N) return;
  int n = i % N; int t = i / N;
  int k = t % K; int lay = t / K;
  Wt[((size_t)lay * N + n) * K + k] = f2b(W[i]);
}

// ------------------------------------------------------------------ GEMM
// C(MxN) = A(MxK, bf16) * W^T(NxK, bf16), f32 accumulate, fused epilogues.
// Double-buffered LDS, one barrier per 32-deep K-step, register-staged loads.
enum { MODE_QKV = 0, MODE_RES = 1, MODE_GELU = 2, MODE_RES_BIAS = 3 };

template <int MODE>
__global__ void __launch_bounds__(256)
gemm_bf16(const u16* __restrict__ A, const u16* __restrict__ Wt,
          const float* __restrict__ res, const float* __restrict__ bias,
          float* __restrict__ outF, u16* __restrict__ outB,
          u16* __restrict__ outB2, u16* __restrict__ outB3,
          int M, int N, int K) {
  __shared__ __align__(16) u16 As[2][128][40];   // +8 pad: 80B row stride
  __shared__ __align__(16) u16 Bs[2][128][40];

  const int tid  = threadIdx.x;
  const int lane = tid & 31, wave = tid >> 5;
  const int wm = wave >> 2, wn = wave & 3;       // 2x4 wave grid, 64x32 per wave
  const int rowBase = blockIdx.y * 128;
  const int colBase = blockIdx.x * 128;
  const int hl = lane >> 4, r = lane & 15;
  const int rr = tid >> 2, g = tid & 3;          // tile-copy coords

  const f32x8 z8 = {0, 0, 0, 0, 0, 0, 0, 0};
  f32x8 acc[4][2];
#pragma unroll
  for (int mt = 0; mt < 4; mt++)
#pragma unroll
    for (int nt = 0; nt < 2; nt++) acc[mt][nt] = z8;

  u32x4 ra0, ra1, rb0, rb1;
  auto loadTile = [&](int kt) {
    ra0 = *(const u32x4*)(A  + (size_t)(rowBase + rr)      * K + kt + g * 8);
    ra1 = *(const u32x4*)(A  + (size_t)(rowBase + rr + 64) * K + kt + g * 8);
    rb0 = *(const u32x4*)(Wt + (size_t)(colBase + rr)      * K + kt + g * 8);
    rb1 = *(const u32x4*)(Wt + (size_t)(colBase + rr + 64) * K + kt + g * 8);
  };
  auto storeTile = [&](int p) {
    *(u32x4*)&As[p][rr][g * 8]      = ra0;
    *(u32x4*)&As[p][rr + 64][g * 8] = ra1;
    *(u32x4*)&Bs[p][rr][g * 8]      = rb0;
    *(u32x4*)&Bs[p][rr + 64][g * 8] = rb1;
  };

  loadTile(0);
  storeTile(0);
  int p = 0;
  for (int kt = 0; kt < K; kt += 32) {
    const bool hasNext = (kt + 32) < K;
    if (hasNext) loadTile(kt + 32);            // overlap with compute below
    __syncthreads();                           // buf[p] stores now visible
    bf16x16 b0 = ldsB(&Bs[p][wn * 32 + r][0], lane);
    bf16x16 b1 = ldsB(&Bs[p][wn * 32 + 16 + r][0], lane);
#pragma unroll
    for (int mt = 0; mt < 4; mt++) {
      bf16x16 a = ldsA(&As[p][wm * 64 + mt * 16 + r][0], lane);
      acc[mt][0] = wmma_bf16(a, b0, acc[mt][0]);
      acc[mt][1] = wmma_bf16(a, b1, acc[mt][1]);
    }
    if (hasNext) storeTile(p ^ 1);             // other buffer: no barrier needed
    p ^= 1;
  }

  // epilogue: C layout -> VGPR j holds (row j + 8*hl, col = ntbase + r)
#pragma unroll
  for (int mt = 0; mt < 4; mt++) {
#pragma unroll
    for (int nt = 0; nt < 2; nt++) {
#pragma unroll
      for (int j = 0; j < 8; j++) {
        int gr = rowBase + wm * 64 + mt * 16 + j + hl * 8;
        int gc = colBase + wn * 32 + nt * 16 + r;
        float v = acc[mt][nt][j];
        if (MODE == MODE_QKV) {
          int which = gc >> 9;                 // uniform per block
          int gcc = gc & 511;
          u16* dst = which == 0 ? outB : (which == 1 ? outB2 : outB3);
          int b = gr >> 10, l = gr & 1023;     // L = 1024
          int h = gcc >> 6, e = gcc & 63;      // DK = 64
          dst[(((size_t)b * Hc + h) * Lc + l) * DKc + e] = f2b(v);
        } else if (MODE == MODE_RES) {
          v += res[(size_t)gr * N + gc];
          outF[(size_t)gr * N + gc] = v;
          outB[(size_t)gr * N + gc] = f2b(v);
        } else if (MODE == MODE_GELU) {
          v += bias[gc];
          v = 0.5f * v * (1.0f + erff(v * 0.70710678118654752f));   // exact gelu
          outB[(size_t)gr * N + gc] = f2b(v);
        } else {  // MODE_RES_BIAS
          v += bias[gc] + res[(size_t)gr * N + gc];
          outF[(size_t)gr * N + gc] = v;
          outB[(size_t)gr * N + gc] = f2b(v);
        }
      }
    }
  }
}

// ------------------------------------------------------------- flash attention
// grid (L/64, B*H), block 128 (4 waves, 16 query rows each). No 1/sqrt(dk) scale.
__global__ void __launch_bounds__(128)
attn_kernel(const u16* __restrict__ Qg, const u16* __restrict__ Kg,
            const u16* __restrict__ Vg, const unsigned char* __restrict__ mask,
            u16* __restrict__ ctxB) {
  __shared__ __align__(16) u16 Qs[64][72];
  __shared__ __align__(16) u16 Ks[64][72];
  __shared__ __align__(16) u16 Vt[64][72];        // Vt[dk][key]
  __shared__ __align__(16) u16 Pl[4][16][72];     // wave-private P tiles

  const int tid = threadIdx.x, lane = tid & 31, wave = tid >> 5;
  const int hl = lane >> 4, r = lane & 15;
  const int bh = blockIdx.y;                      // b*H + h
  const int b = bh >> 3, h = bh & 7;
  const int qbase = blockIdx.x * 64;
  const size_t base = (size_t)bh * Lc * DKc;
  const float NEGF = -3.0e38f;

  // load Q tile (64x64 bf16)
#pragma unroll
  for (int i = 0; i < 4; i++) {
    int c = tid + i * 128;
    int rr = c >> 3, g = c & 7;
    *(u32x4*)&Qs[rr][g * 8] =
        *(const u32x4*)(Qg + base + (size_t)(qbase + rr) * DKc + g * 8);
  }
  __syncthreads();

  bf16x16 qa0 = ldsA(&Qs[wave * 16 + r][0], lane);
  bf16x16 qa1 = ldsA(&Qs[wave * 16 + r][32], lane);

  const f32x8 z8 = {0, 0, 0, 0, 0, 0, 0, 0};
  f32x8 o[4];
  float mst[8], lst[8], mq[8];
#pragma unroll
  for (int nt = 0; nt < 4; nt++) o[nt] = z8;
#pragma unroll
  for (int j = 0; j < 8; j++) {
    mst[j] = NEGF; lst[j] = 0.0f;
    int qr = qbase + wave * 16 + j + hl * 8;
    mq[j] = mask[b * Lc + qr] ? 0.0f : 1.0f;      // 1 => invalid query
  }

  for (int kb = 0; kb < Lc; kb += 64) {
    __syncthreads();
#pragma unroll
    for (int i = 0; i < 4; i++) {
      int c = tid + i * 128;
      int rr = c >> 3, g = c & 7;
      *(u32x4*)&Ks[rr][g * 8] =
          *(const u32x4*)(Kg + base + (size_t)(kb + rr) * DKc + g * 8);
      u32x4 vv = *(const u32x4*)(Vg + base + (size_t)(kb + rr) * DKc + g * 8);
#pragma unroll
      for (int t = 0; t < 4; t++) {               // transpose V into LDS
        unsigned int w = vv[t];
        Vt[g * 8 + t * 2 + 0][rr] = (u16)(w & 0xFFFFu);
        Vt[g * 8 + t * 2 + 1][rr] = (u16)(w >> 16);
      }
    }
    __syncthreads();

    // S = Q K^T  (16 queries x 64 keys)
    f32x8 s[4];
#pragma unroll
    for (int nt = 0; nt < 4; nt++) {
      bf16x16 kb0 = ldsB(&Ks[nt * 16 + r][0], lane);
      bf16x16 kb1 = ldsB(&Ks[nt * 16 + r][32], lane);
      f32x8 c = z8;
      c = wmma_bf16(qa0, kb0, c);
      c = wmma_bf16(qa1, kb1, c);
      s[nt] = c;
    }

    // additive mask (finfo.min where query or key invalid)
    float mk[4];
#pragma unroll
    for (int nt = 0; nt < 4; nt++)
      mk[nt] = mask[b * Lc + kb + nt * 16 + r] ? 0.0f : 1.0f;
#pragma unroll
    for (int nt = 0; nt < 4; nt++)
#pragma unroll
      for (int j = 0; j < 8; j++)
        s[nt][j] += (mq[j] + mk[nt] > 0.0f) ? NEGF : 0.0f;

    // online softmax (rows live in one VGPR across a 16-lane half)
    float scale[8];
#pragma unroll
    for (int j = 0; j < 8; j++) {
      float m = fmaxf(fmaxf(s[0][j], s[1][j]), fmaxf(s[2][j], s[3][j]));
      m = fmaxf(m, __shfl_xor(m, 1, 32));
      m = fmaxf(m, __shfl_xor(m, 2, 32));
      m = fmaxf(m, __shfl_xor(m, 4, 32));
      m = fmaxf(m, __shfl_xor(m, 8, 32));
      float mn = fmaxf(mst[j], m);
      scale[j] = __expf(mst[j] - mn);
      mst[j] = mn;
    }
#pragma unroll
    for (int j = 0; j < 8; j++) {
      float ssum = 0.0f;
#pragma unroll
      for (int nt = 0; nt < 4; nt++) {
        float pp = __expf(s[nt][j] - mst[j]);
        s[nt][j] = pp;
        ssum += pp;
      }
      ssum += __shfl_xor(ssum, 1, 32);
      ssum += __shfl_xor(ssum, 2, 32);
      ssum += __shfl_xor(ssum, 4, 32);
      ssum += __shfl_xor(ssum, 8, 32);
      lst[j] = lst[j] * scale[j] + ssum;
#pragma unroll
      for (int nt = 0; nt < 4; nt++) o[nt][j] *= scale[j];
    }

    // C-layout P -> A-layout via wave-private LDS round trip
#pragma unroll
    for (int nt = 0; nt < 4; nt++)
#pragma unroll
      for (int j = 0; j < 8; j++)
        Pl[wave][j + hl * 8][nt * 16 + r] = f2b(s[nt][j]);

    bf16x16 pa0 = ldsA(&Pl[wave][r][0], lane);
    bf16x16 pa1 = ldsA(&Pl[wave][r][32], lane);
#pragma unroll
    for (int nt = 0; nt < 4; nt++) {              // O += P * V
      bf16x16 vb0 = ldsB(&Vt[nt * 16 + r][0], lane);
      bf16x16 vb1 = ldsB(&Vt[nt * 16 + r][32], lane);
      o[nt] = wmma_bf16(pa0, vb0, o[nt]);
      o[nt] = wmma_bf16(pa1, vb1, o[nt]);
    }
  }

  // normalize and emit ctx (B, L, H*DK) in bf16 for the output projection
#pragma unroll
  for (int j = 0; j < 8; j++) {
    float inv = 1.0f / lst[j];
    int qr = qbase + wave * 16 + j + hl * 8;
#pragma unroll
    for (int nt = 0; nt < 4; nt++) {
      float val = o[nt][j] * inv;
      ctxB[((size_t)b * Lc + qr) * (Hc * DKc) + h * 64 + nt * 16 + r] = f2b(val);
    }
  }
}

// -------------------------------------------------------------------- host
extern "C" void kernel_launch(void* const* d_in, const int* in_sizes, int n_in,
                              void* d_out, int out_size, void* d_ws, size_t ws_size,
                              hipStream_t stream) {
  (void)in_sizes; (void)n_in; (void)out_size; (void)ws_size;
  const float* x          = (const float*)d_in[0];
  const unsigned char* mk = (const unsigned char*)d_in[1];
  const float* Wq = (const float*)d_in[2];
  const float* Wk = (const float*)d_in[3];
  const float* Wv = (const float*)d_in[4];
  const float* Wo = (const float*)d_in[5];
  const float* W1 = (const float*)d_in[6];
  const float* b1 = (const float*)d_in[7];
  const float* W2 = (const float*)d_in[8];
  const float* b2 = (const float*)d_in[9];

  char* ws = (char*)d_ws;
  size_t off = 0;
  auto alloc = [&](size_t bytes) -> char* {
    char* p = ws + off;
    off += (bytes + 255) & ~(size_t)255;
    return p;
  };
  u16* WqkvT = (u16*)alloc((size_t)NLAY * 1536 * 512 * 2);  // fused q|k|v bands
  u16* WoT   = (u16*)alloc((size_t)NLAY * 512 * 512 * 2);
  u16* W1T   = (u16*)alloc((size_t)NLAY * 1024 * 512 * 2);
  u16* W2T   = (u16*)alloc((size_t)NLAY * 512 * 1024 * 2);
  u16*   xb   = (u16*)alloc((size_t)MR * 512 * 2);
  float* xf   = (float*)alloc((size_t)MR * 512 * 4);
  float* xmf  = (float*)alloc((size_t)MR * 512 * 4);
  u16*   xmb  = (u16*)alloc((size_t)MR * 512 * 2);
  u16*   fb   = (u16*)alloc((size_t)MR * 1024 * 2);
  u16*   qb   = (u16*)alloc((size_t)MR * 512 * 2);
  u16*   kbuf = (u16*)alloc((size_t)MR * 512 * 2);
  u16*   vb   = (u16*)alloc((size_t)MR * 512 * 2);
  u16*   ctxb = (u16*)alloc((size_t)MR * 512 * 2);

  // one-time weight conversion / transposition to bf16
  {
    int t = NLAY * Hc * Dc * DKc;
    cvt_wqkvT<<<(t + 255) / 256, 256, 0, stream>>>(Wq, WqkvT, 0);
    cvt_wqkvT<<<(t + 255) / 256, 256, 0, stream>>>(Wk, WqkvT, 512);
    cvt_wqkvT<<<(t + 255) / 256, 256, 0, stream>>>(Wv, WqkvT, 1024);
    int to = NLAY * 512 * 512;
    cvt_wT<<<(to + 255) / 256, 256, 0, stream>>>(Wo, WoT, 512, 512);
    int t1 = NLAY * 512 * 1024;
    cvt_wT<<<(t1 + 255) / 256, 256, 0, stream>>>(W1, W1T, 512, 1024);
    cvt_wT<<<(t1 + 255) / 256, 256, 0, stream>>>(W2, W2T, 1024, 512);
    int tx = MR * 512;
    cvt_bf16<<<(tx + 255) / 256, 256, 0, stream>>>(x, xb, tx);
  }

  const dim3 gQKV(1536 / 128, MR / 128);    // (12, 64)
  const dim3 g512(512 / 128, MR / 128);     // (4, 64)
  const dim3 g1024(1024 / 128, MR / 128);   // (8, 64)
  const dim3 ga(Lc / 64, Bc * Hc);          // (16, 64)

  const float* curF = x;                    // f32 residual source
  for (int lay = 0; lay < NLAY; lay++) {
    const u16* wqkv = WqkvT + (size_t)lay * 1536 * 512;
    const u16* wo   = WoT + (size_t)lay * 512 * 512;
    const u16* w1   = W1T + (size_t)lay * 1024 * 512;
    const u16* w2   = W2T + (size_t)lay * 512 * 1024;
    const float* bias1 = b1 + (size_t)lay * DFFc;
    const float* bias2 = b2 + (size_t)lay * Dc;

    gemm_bf16<MODE_QKV><<<gQKV, 256, 0, stream>>>(
        xb, wqkv, nullptr, nullptr, nullptr, qb, kbuf, vb, MR, 1536, 512);
    attn_kernel<<<ga, 128, 0, stream>>>(qb, kbuf, vb, mk, ctxb);
    gemm_bf16<MODE_RES><<<g512, 256, 0, stream>>>(
        ctxb, wo, curF, nullptr, xmf, xmb, nullptr, nullptr, MR, 512, 512);
    gemm_bf16<MODE_GELU><<<g1024, 256, 0, stream>>>(
        xmb, w1, nullptr, bias1, nullptr, fb, nullptr, nullptr, MR, 1024, 512);
    float* outF = (lay == NLAY - 1) ? (float*)d_out : xf;
    gemm_bf16<MODE_RES_BIAS><<<g512, 256, 0, stream>>>(
        fb, w2, xmf, bias2, outF, xb, nullptr, nullptr, MR, 512, 1024);
    curF = outF;
  }
}
